// MixtureOfExperts_82669530514267
// MI455X (gfx1250) — matrix operand order repaired and checked by reading, not compiled, not run on hardware
//
#include <hip/hip_runtime.h>
#include <hip/hip_bf16.h>

// ---------------------------------------------------------------------------
// MoE (top-2 of 8 experts), H=1024, FF=2048, T=4096 tokens, fp32 in/out.
// Expert-major grouped GEMM, bf16 WMMA (v_wmma_f32_16x16x32_bf16), fused
// FFN with hmid tile resident in LDS (CDNA5 320KB LDS/WGP).
// Weights are pre-converted once per launch to bf16, N-major (Wt[n][k]) so
// B fragments are two contiguous b128 loads per lane.
// ---------------------------------------------------------------------------

#define T_TOK   4096
#define HDIM    1024
#define FFDIM   2048
#define NEXP    8
#define WAVES   8          // 256 threads / wave32
#define MTILE   32         // tokens per workgroup (2 x 16-row WMMA tiles)

typedef __attribute__((ext_vector_type(16))) __bf16 v16bf;
typedef __attribute__((ext_vector_type(8)))  float  v8f;

static __device__ __forceinline__ __bf16 f2bf(float f) { return (__bf16)f; }

union ABFrag { v16bf v; uint4 q[2]; };

// A fragment (16x32 MxK bf16) from an LDS tile, ISA wave32 layout:
//   lane L holds row m = L%16; elements e=0..7  -> k = k0 + (L/16)*8 + e,
//   elements e=8..15 -> k = k0 + 16 + (L/16)*8 + (e-8).  Two 16B chunks.
static __device__ __forceinline__ v16bf load_a_lds(const __bf16* base, int stride,
                                                   int k0, int lane) {
    const int m  = lane & 15;
    const int kg = lane >> 4;
    const __bf16* p = base + m * stride + k0 + kg * 8;
    ABFrag u;
    u.q[0] = *reinterpret_cast<const uint4*>(p);        // k0 .. k0+7
    u.q[1] = *reinterpret_cast<const uint4*>(p + 16);   // k0+16 .. k0+23
    return u.v;
}

// B fragment (32x16 KxN bf16) from N-major bf16 weights Wt[n][k] (ldk = K dim):
//   lane L holds column n = L%16; e=0..7 -> k = k0 + (L/16)*8 + e,
//   e=8..15 -> k = k0 + 16 + (L/16)*8 + (e-8)  ->  two b128 loads per lane.
static __device__ __forceinline__ v16bf load_b_nmajor(const __bf16* Wt, int ldk,
                                                      int k0, int n0, int lane) {
    const int nl = lane & 15;
    const int kg = lane >> 4;
    const __bf16* p = Wt + (size_t)(n0 + nl) * ldk + k0 + kg * 8;
    ABFrag u;
    u.q[0] = *reinterpret_cast<const uint4*>(p);        // k .. k+7
    u.q[1] = *reinterpret_cast<const uint4*>(p + 16);   // k+16 .. k+23
    return u.v;
}

// ---------------------------------------------------------------------------
// Kernel 0: zero the output accumulator and the expert counters.
// ---------------------------------------------------------------------------
__global__ void moe_zero_kernel(float* __restrict__ out, int* __restrict__ cnt) {
    const int i = blockIdx.x * blockDim.x + threadIdx.x;
    if (i < T_TOK * HDIM) out[i] = 0.0f;
    if (i < NEXP)         cnt[i] = 0;
}

// ---------------------------------------------------------------------------
// Kernel 1: weight convert + transpose, fp32 [E][R][C] -> bf16 [E][C][R].
// For W1 (R=H, C=FF): W1t[e][f][h];  for W2 (R=FF, C=H): W2t[e][h][f].
// ---------------------------------------------------------------------------
__global__ void moe_wconv_kernel(const float* __restrict__ src,
                                 __bf16* __restrict__ dst,
                                 int R, int C) {
    const int e = blockIdx.y;
    const size_t plane = (size_t)R * C;
    const float* s = src + (size_t)e * plane;
    __bf16*      d = dst + (size_t)e * plane;
    // i indexes the *destination* linearly (c major, r fastest): coalesced writes
    for (size_t i = (size_t)blockIdx.x * blockDim.x + threadIdx.x;
         i < plane; i += (size_t)gridDim.x * blockDim.x) {
        const int r = (int)(i % R);
        const int c = (int)(i / R);
        d[i] = f2bf(s[(size_t)r * C + c]);
    }
}

// ---------------------------------------------------------------------------
// Kernel 2: gating — logits, softmax, top-2, renormalize, scatter to
// per-expert token lists (expert-major grouped layout).
// ---------------------------------------------------------------------------
__global__ void moe_gate_kernel(const float* __restrict__ x,
                                const float* __restrict__ Wg,
                                const float* __restrict__ bg,
                                int*   __restrict__ cnt,
                                int*   __restrict__ tokidx,
                                float* __restrict__ tokw) {
    const int t = blockIdx.x * blockDim.x + threadIdx.x;
    if (t >= T_TOK) return;

    float acc[NEXP];
#pragma unroll
    for (int e = 0; e < NEXP; ++e) acc[e] = bg[e];

    const float* xr = x + (size_t)t * HDIM;
    for (int h = 0; h < HDIM; ++h) {
        const float xv = xr[h];
        const float* wr = Wg + (size_t)h * NEXP;
#pragma unroll
        for (int e = 0; e < NEXP; ++e) acc[e] += xv * wr[e];
    }

    float mx = acc[0];
#pragma unroll
    for (int e = 1; e < NEXP; ++e) mx = fmaxf(mx, acc[e]);
    float p[NEXP], s = 0.0f;
#pragma unroll
    for (int e = 0; e < NEXP; ++e) { p[e] = __expf(acc[e] - mx); s += p[e]; }
    const float inv = 1.0f / s;
#pragma unroll
    for (int e = 0; e < NEXP; ++e) p[e] *= inv;

    int i1 = 0; float p1 = p[0];
#pragma unroll
    for (int e = 1; e < NEXP; ++e) if (p[e] > p1) { p1 = p[e]; i1 = e; }
    int i2 = -1; float p2 = -1.0f;
#pragma unroll
    for (int e = 0; e < NEXP; ++e) if (e != i1 && p[e] > p2) { p2 = p[e]; i2 = e; }

    const float denom = p1 + p2;
    const float w1 = p1 / denom;
    const float w2 = p2 / denom;

    int pos = atomicAdd(&cnt[i1], 1);
    tokidx[i1 * T_TOK + pos] = t;  tokw[i1 * T_TOK + pos] = w1;
    pos = atomicAdd(&cnt[i2], 1);
    tokidx[i2 * T_TOK + pos] = t;  tokw[i2 * T_TOK + pos] = w2;
}

// ---------------------------------------------------------------------------
// Kernel 3: fused expert FFN.  grid = (T/32 tiles, NEXP experts),
// block = 256 (8 wave32).  xs (32xH bf16, 64KB) and hm (32xFF bf16, 128KB)
// live in dynamic LDS.  GEMM1: hm = relu(xs @ W1 + b1) with W1t bf16 N-major;
// GEMM2: atomically accumulate w * (hm @ W2 + b2) into out.
// Each B fragment feeds two 16-row WMMA tiles (MTILE=32).
// ---------------------------------------------------------------------------
__global__ void __launch_bounds__(256)
moe_expert_kernel(const float*  __restrict__ x,
                  const __bf16* __restrict__ W1t,   // [E][FF][H]  (n-major)
                  const float*  __restrict__ b1,
                  const __bf16* __restrict__ W2t,   // [E][H][FF]  (n-major)
                  const float*  __restrict__ b2,
                  const int*    __restrict__ cnt,
                  const int*    __restrict__ tokidx,
                  const float*  __restrict__ tokw,
                  float*        __restrict__ out) {
    const int e     = blockIdx.y;
    const int row0  = blockIdx.x * MTILE;
    const int n_tok = cnt[e];
    if (row0 >= n_tok) return;          // past this expert's token count

    extern __shared__ __bf16 smem[];
    __bf16* xs = smem;                  // [MTILE][HDIM]
    __bf16* hm = smem + MTILE * HDIM;   // [MTILE][FFDIM]
    __shared__ int   tk[MTILE];
    __shared__ float wt[MTILE];

    const int tid  = threadIdx.x;
    const int lane = tid & 31;
    const int wv   = tid >> 5;

    if (tid < MTILE) {
        const int idx = row0 + tid;
        if (idx < n_tok) {
            tk[tid] = tokidx[e * T_TOK + idx];
            wt[tid] = tokw[e * T_TOK + idx];
        } else {                         // pad: valid row, zero weight
            tk[tid] = tokidx[e * T_TOK + row0];
            wt[tid] = 0.0f;
        }
    }
    __syncthreads();

    // Stage gathered token rows fp32 -> bf16 into LDS (coalesced).
    for (int i = tid; i < MTILE * HDIM; i += blockDim.x) {
        const int r = i >> 10;           // HDIM == 1024
        const int c = i & (HDIM - 1);
        xs[r * HDIM + c] = f2bf(x[(size_t)tk[r] * HDIM + c]);
    }
    __syncthreads();

    const __bf16* W1e = W1t + (size_t)e * FFDIM * HDIM;
    const float*  b1e = b1 + e * FFDIM;
    const int nl = lane & 15;
    const int mg = lane >> 4;

    // ---- GEMM1: [32 x FF] = xs[32 x H] @ W1 ---------------------------------
    for (int nt = 0; nt < (FFDIM / WAVES) / 16; ++nt) {      // 16 N-tiles/wave
        const int n0 = wv * (FFDIM / WAVES) + nt * 16;
        v8f c0 = {0.f, 0.f, 0.f, 0.f, 0.f, 0.f, 0.f, 0.f};
        v8f c1 = c0;
        for (int k0 = 0; k0 < HDIM; k0 += 32) {
            const v16bf b  = load_b_nmajor(W1e, HDIM, k0, n0, lane);
            const v16bf a0 = load_a_lds(xs,               HDIM, k0, lane);
            const v16bf a1 = load_a_lds(xs + 16 * HDIM,   HDIM, k0, lane);
            if (k0 + 32 < HDIM)
                __builtin_prefetch(&W1e[(size_t)(n0 + nl) * HDIM + k0 + 32], 0, 1);
            c0 = __builtin_amdgcn_wmma_f32_16x16x32_bf16(
                     false, a0, false, b, (short)0, c0, false, false);
            c1 = __builtin_amdgcn_wmma_f32_16x16x32_bf16(
                     false, a1, false, b, (short)0, c1, false, false);
        }
#pragma unroll
        for (int j = 0; j < 8; ++j) {
            const int m = mg * 8 + j;
            const int n = n0 + nl;
            const float bias = b1e[n];
            const float h0 = c0[j] + bias;
            const float h1 = c1[j] + bias;
            hm[m * FFDIM + n]        = f2bf(h0 > 0.0f ? h0 : 0.0f);
            hm[(16 + m) * FFDIM + n] = f2bf(h1 > 0.0f ? h1 : 0.0f);
        }
    }
    __syncthreads();

    const __bf16* W2e = W2t + (size_t)e * HDIM * FFDIM;
    const float*  b2e = b2 + e * HDIM;

    // ---- GEMM2: [32 x H] = hm[32 x FF] @ W2, weighted atomic add ------------
    for (int nt = 0; nt < (HDIM / WAVES) / 16; ++nt) {       // 8 N-tiles/wave
        const int n0 = wv * (HDIM / WAVES) + nt * 16;
        v8f c0 = {0.f, 0.f, 0.f, 0.f, 0.f, 0.f, 0.f, 0.f};
        v8f c1 = c0;
        for (int k0 = 0; k0 < FFDIM; k0 += 32) {
            const v16bf b  = load_b_nmajor(W2e, FFDIM, k0, n0, lane);
            const v16bf a0 = load_a_lds(hm,              FFDIM, k0, lane);
            const v16bf a1 = load_a_lds(hm + 16 * FFDIM, FFDIM, k0, lane);
            if (k0 + 32 < FFDIM)
                __builtin_prefetch(&W2e[(size_t)(n0 + nl) * FFDIM + k0 + 32], 0, 1);
            c0 = __builtin_amdgcn_wmma_f32_16x16x32_bf16(
                     false, a0, false, b, (short)0, c0, false, false);
            c1 = __builtin_amdgcn_wmma_f32_16x16x32_bf16(
                     false, a1, false, b, (short)0, c1, false, false);
        }
#pragma unroll
        for (int j = 0; j < 8; ++j) {
            const int m = mg * 8 + j;
            const int n = n0 + nl;
            const float bias = b2e[n];
            const float w0 = wt[m];
            const float w1 = wt[16 + m];
            if (w0 != 0.0f)
                unsafeAtomicAdd(&out[(size_t)tk[m] * HDIM + n],      w0 * (c0[j] + bias));
            if (w1 != 0.0f)
                unsafeAtomicAdd(&out[(size_t)tk[16 + m] * HDIM + n], w1 * (c1[j] + bias));
        }
    }
}

// ---------------------------------------------------------------------------
// Host launcher.
// Workspace layout (aligned):
//   cnt[8] | tokidx[8*T] | tokw[8*T] | W1t bf16 [E][FF][H] | W2t bf16 [E][H][FF]
//   ~0.26 MB routing + 67 MB bf16 weights.
// ---------------------------------------------------------------------------
extern "C" void kernel_launch(void* const* d_in, const int* in_sizes, int n_in,
                              void* d_out, int out_size, void* d_ws, size_t ws_size,
                              hipStream_t stream) {
    const float* x  = (const float*)d_in[0];
    const float* Wg = (const float*)d_in[1];
    const float* bg = (const float*)d_in[2];
    const float* W1 = (const float*)d_in[3];
    const float* b1 = (const float*)d_in[4];
    const float* W2 = (const float*)d_in[5];
    const float* b2 = (const float*)d_in[6];
    float* out = (float*)d_out;

    char* ws = (char*)d_ws;
    int*    cnt    = (int*)ws;                          ws += 256;
    int*    tokidx = (int*)ws;                          ws += (size_t)NEXP * T_TOK * sizeof(int);
    float*  tokw   = (float*)ws;                        ws += (size_t)NEXP * T_TOK * sizeof(float);
    __bf16* W1t    = (__bf16*)ws;                       ws += (size_t)NEXP * FFDIM * HDIM * sizeof(__bf16);
    __bf16* W2t    = (__bf16*)ws;

    // 0) zero output + counters
    {
        const int n = T_TOK * HDIM;
        moe_zero_kernel<<<(n + 255) / 256, 256, 0, stream>>>(out, cnt);
    }
    // 1) weight convert+transpose fp32 -> bf16 N-major
    {
        dim3 grid(2048, NEXP);
        moe_wconv_kernel<<<grid, 256, 0, stream>>>(W1, W1t, HDIM, FFDIM);
        moe_wconv_kernel<<<grid, 256, 0, stream>>>(W2, W2t, FFDIM, HDIM);
    }
    // 2) gating + expert-major scatter
    moe_gate_kernel<<<(T_TOK + 255) / 256, 256, 0, stream>>>(
        x, Wg, bg, cnt, tokidx, tokw);
    // 3) fused expert FFN (WMMA)
    {
        dim3 grid(T_TOK / MTILE, NEXP);
        const size_t lds = (size_t)(MTILE * HDIM + MTILE * FFDIM) * sizeof(__bf16);
        moe_expert_kernel<<<grid, 256, lds, stream>>>(
            x, W1t, b1, W2t, b2, cnt, tokidx, tokw, out);
    }
}
